// TransformerBlock_8675833938648
// MI455X (gfx1250) — matrix operand order repaired and checked by reading, not compiled, not run on hardware
//
#include <hip/hip_runtime.h>

typedef __attribute__((ext_vector_type(16))) __bf16 v16bf;
typedef __attribute__((ext_vector_type(8)))  __bf16 v8bf;
typedef __attribute__((ext_vector_type(8)))  float  v8f;

#define CAT16(lo, hi) __builtin_shufflevector((lo), (hi), 0,1,2,3,4,5,6,7,8,9,10,11,12,13,14,15)

constexpr int Bc  = 2;
constexpr int Sc  = 2048;
constexpr int Ec  = 1024;
constexpr int Hc  = 16;
constexpr int DHc = 64;
constexpr int FFc = 4096;
constexpr int BSc = Bc * Sc;
constexpr int KSTEP = 32;
constexpr int LDSTR = KSTEP + 8;   // padded LDS row stride (40 bf16 = 80 B)

#if __has_builtin(__builtin_amdgcn_tensor_load_to_lds) && __has_builtin(__builtin_amdgcn_s_wait_tensorcnt)
#define USE_TDM 1
#endif

__device__ __forceinline__ __bf16 f2bf(float f) {
  unsigned int u = __builtin_bit_cast(unsigned int, f);
  u += 0x7fffu + ((u >> 16) & 1u);                 // round-to-nearest-even
  unsigned short h = (unsigned short)(u >> 16);
  return __builtin_bit_cast(__bf16, h);
}

#ifdef USE_TDM
// Issue a 2D TDM tile load: tile (rows x 32 bf16) from a row-major matrix
// (strideElems per row) into LDS at byte offset ldsOff, with LDS padding of
// 4 DWORDs after every 16 DWORDs (=> stride-40 bf16 rows, matching LDSTR).
// 6-arg builtin form (clang-23 / therock headers): (v4u, v8i, v4i, v4i, v8i, i32)
__device__ __forceinline__ void tdm_load_tile(const __bf16* gsrc, unsigned ldsOff,
                                              int rows, int strideElems,
                                              int dim0, int dim1) {
  typedef __attribute__((ext_vector_type(4))) unsigned int tdm_v4u;
  typedef __attribute__((ext_vector_type(8))) int          tdm_v8i;
  typedef __attribute__((ext_vector_type(4))) int          tdm_v4i;
  unsigned long ga = (unsigned long)gsrc;
  tdm_v4u g0;
  g0[0] = 1u;                                            // count=1, user mode
  g0[1] = ldsOff;                                        // lds_addr
  g0[2] = (unsigned)(ga & 0xffffffffu);                  // global_addr[31:0]
  g0[3] = (unsigned)((ga >> 32) & 0x1ffffffu) | (2u << 30); // ga[56:32] | type=2
  tdm_v8i g1;
  // data_size=1(2B) | pad_enable | pad_interval=3(16 DW) | pad_amount=3(4 DW)
  g1[0] = (int)((1u << 16) | (1u << 20) | (3u << 22) | (3u << 25));
  g1[1] = (int)(((unsigned)dim0 & 0xffffu) << 16);                 // tensor_dim0 lo
  g1[2] = (int)(((unsigned)dim0 >> 16) | (((unsigned)dim1 & 0xffffu) << 16));
  g1[3] = (int)(((unsigned)dim1 >> 16) | ((unsigned)KSTEP << 16)); // tile_dim0=32
  g1[4] = rows;                                                    // tile_dim1
  g1[5] = strideElems;                                             // dim0_stride lo
  g1[6] = 0;
  g1[7] = 0;
  tdm_v4i z4 = {0, 0, 0, 0};
  tdm_v8i z8 = {0, 0, 0, 0, 0, 0, 0, 0};
  __builtin_amdgcn_tensor_load_to_lds(g0, g1, z4, z4, z8, 0);
}
#endif

// ---------------------------------------------------------------------------
// fp32 -> bf16 elementwise (weights)
// ---------------------------------------------------------------------------
__global__ __launch_bounds__(256)
void f32_to_bf16_kernel(const float* __restrict__ in, __bf16* __restrict__ out, long n) {
  long i = ((long)blockIdx.x * 256 + threadIdx.x) * 4;
  if (i + 3 < n) {
    float4 v = *(const float4*)(in + i);
    out[i + 0] = f2bf(v.x);
    out[i + 1] = f2bf(v.y);
    out[i + 2] = f2bf(v.z);
    out[i + 3] = f2bf(v.w);
  }
}

// ---------------------------------------------------------------------------
// LayerNorm over E=1024, one row per 256-thread block (8 waves, wave32)
// ---------------------------------------------------------------------------
__global__ __launch_bounds__(256)
void layernorm_kernel(const float* __restrict__ x, const float* __restrict__ g,
                      const float* __restrict__ be, __bf16* __restrict__ out) {
  __shared__ float redS[8], redS2[8];
  const long row = blockIdx.x;
  const float4 xv = ((const float4*)(x + row * Ec))[threadIdx.x];
  float s  = xv.x + xv.y + xv.z + xv.w;
  float s2 = xv.x * xv.x + xv.y * xv.y + xv.z * xv.z + xv.w * xv.w;
#pragma unroll
  for (int off = 16; off >= 1; off >>= 1) {
    s  += __shfl_xor(s, off, 32);
    s2 += __shfl_xor(s2, off, 32);
  }
  if ((threadIdx.x & 31) == 0) {
    redS[threadIdx.x >> 5]  = s;
    redS2[threadIdx.x >> 5] = s2;
  }
  __syncthreads();
  float ts = 0.f, ts2 = 0.f;
#pragma unroll
  for (int i = 0; i < 8; ++i) { ts += redS[i]; ts2 += redS2[i]; }
  const float mean = ts * (1.f / Ec);
  const float var  = ts2 * (1.f / Ec) - mean * mean;
  const float rstd = rsqrtf(var + 1e-5f);
  const int c = threadIdx.x * 4;
  out[row * Ec + c + 0] = f2bf((xv.x - mean) * rstd * g[c + 0] + be[c + 0]);
  out[row * Ec + c + 1] = f2bf((xv.y - mean) * rstd * g[c + 1] + be[c + 1]);
  out[row * Ec + c + 2] = f2bf((xv.z - mean) * rstd * g[c + 2] + be[c + 2]);
  out[row * Ec + c + 3] = f2bf((xv.w - mean) * rstd * g[c + 3] + be[c + 3]);
}

// ---------------------------------------------------------------------------
// bf16 WMMA GEMM, 8 waves (2x4), K-step 32, double-buffered LDS.
// A tile loaded by TDM (if available) else staged through VGPRs; B tile
// staged through VGPRs and stored transposed for contiguous B-fragments.
// Dynamic LDS layout (byte offsets, base assumed 0 for TDM lds_addr):
//   [0, ABYTES)        As buf0      [ABYTES, 2*ABYTES)      As buf1
//   [2*ABYTES, +BBY)   Bs buf0      [...]                   Bs buf1
// ---------------------------------------------------------------------------
template <int BM, int BN>
__global__ __launch_bounds__(256)
void gemm_bf16(const __bf16* __restrict__ Ag, int lda, long aBatch,
               const __bf16* __restrict__ Bg, int ldb, long bBatch,
               const float* __restrict__ bias,
               const float* __restrict__ resid,
               float* __restrict__ outF,
               __bf16* __restrict__ outH,
               int ldc, long cBatch,
               int K, int headsH, int relu, int Mtot) {
  constexpr int MI = BM / 32;
  constexpr int NI = BN / 64;
  constexpr unsigned AWORDS = (unsigned)BM * LDSTR;   // bf16 units per A buffer
  constexpr unsigned BWORDS = (unsigned)BN * LDSTR;
  constexpr unsigned ABYTES = AWORDS * 2;
  extern __shared__ __bf16 smem[];

  const int z = blockIdx.z;
  const __bf16* A  = Ag + (long)(z / headsH) * aBatch;
  const __bf16* Bm = Bg + (long)(z % headsH) * bBatch;
  const long cOff  = (long)z * cBatch;

  const int m0 = blockIdx.y * BM;
  const int n0 = blockIdx.x * BN;

  const int tid   = threadIdx.x;
  const int lane  = tid & 31;
  const int wave  = tid >> 5;
  const int mBase = (wave >> 2) * (BM / 2);
  const int nBase = (wave & 3) * (BN / 4);
  const int hl = lane >> 4;
  const int nc = lane & 15;

  v8f acc[MI][NI];
#pragma unroll
  for (int i = 0; i < MI; ++i)
#pragma unroll
    for (int j = 0; j < NI; ++j) acc[i][j] = (v8f){0, 0, 0, 0, 0, 0, 0, 0};

  constexpr int TPR = 256 / BM;                  // threads per A row
  constexpr int ACH = KSTEP / TPR / 8;           // A v8bf chunks per thread
  const int arow  = tid / TPR;
  const int acol0 = (tid % TPR) * (KSTEP / TPR);
  (void)arow; (void)acol0;

  v8bf aReg[ACH];
  v8bf bReg[BN / 64];

  // ---------------- prologue: tile 0 ----------------
#ifdef USE_TDM
  if (tid < 32) tdm_load_tile(A + (long)m0 * lda, 0u, BM, lda, K, Mtot);
#else
#pragma unroll
  for (int c = 0; c < ACH; ++c)
    aReg[c] = *(const v8bf*)(A + (long)(m0 + arow) * lda + acol0 + c * 8);
#endif
#pragma unroll
  for (int it = 0; it < (BN / 64); ++it) {
    int chunk = tid + it * 256;
    int r = chunk / (BN / 8);
    int c = (chunk % (BN / 8)) * 8;
    bReg[it] = *(const v8bf*)(Bm + (long)r * ldb + n0 + c);
  }
#ifndef USE_TDM
#pragma unroll
  for (int c = 0; c < ACH; ++c)
    *(v8bf*)&smem[(unsigned)arow * LDSTR + acol0 + c * 8] = aReg[c];
#endif
#pragma unroll
  for (int it = 0; it < (BN / 64); ++it) {
    int chunk = tid + it * 256;
    int r = chunk / (BN / 8);
    int c = (chunk % (BN / 8)) * 8;
#pragma unroll
    for (int e = 0; e < 8; ++e) smem[2 * AWORDS + (unsigned)(c + e) * LDSTR + r] = bReg[it][e];
  }
#ifdef USE_TDM
  if (tid < 32) __builtin_amdgcn_s_wait_tensorcnt(0);
#endif
  __syncthreads();

  // ---------------- main loop ----------------
  int buf = 0;
  for (int kt = 0; kt < K; kt += KSTEP, buf ^= 1) {
    const bool more = (kt + KSTEP) < K;
    // issue next tile's global traffic first
    if (more) {
#ifdef USE_TDM
      if (tid < 32)
        tdm_load_tile(A + (long)m0 * lda + kt + KSTEP, (unsigned)(buf ^ 1) * ABYTES,
                      BM, lda, K, Mtot);
#else
#pragma unroll
      for (int c = 0; c < ACH; ++c)
        aReg[c] = *(const v8bf*)(A + (long)(m0 + arow) * lda + kt + KSTEP + acol0 + c * 8);
#endif
#pragma unroll
      for (int it = 0; it < (BN / 64); ++it) {
        int chunk = tid + it * 256;
        int r = chunk / (BN / 8);
        int c = (chunk % (BN / 8)) * 8;
        bReg[it] = *(const v8bf*)(Bm + (long)(kt + KSTEP + r) * ldb + n0 + c);
      }
    }

    const __bf16* Ab = smem + (unsigned)buf * AWORDS;
    const __bf16* Bb = smem + 2 * AWORDS + (unsigned)buf * BWORDS;

    // preload ALL fragments, then burst the WMMAs
    v16bf bfr[NI];
#pragma unroll
    for (int j = 0; j < NI; ++j) {
      const __bf16* p = &Bb[(unsigned)(nBase + j * 16 + nc) * LDSTR + (hl << 3)];
      v8bf lo = *(const v8bf*)p;
      v8bf hi = *(const v8bf*)(p + 16);
      bfr[j] = CAT16(lo, hi);
    }
    v16bf af[MI];
#pragma unroll
    for (int i = 0; i < MI; ++i) {
      const __bf16* p = &Ab[(unsigned)(mBase + i * 16 + nc) * LDSTR + (hl << 3)];
      v8bf lo = *(const v8bf*)p;          // k = base..base+7
      v8bf hi = *(const v8bf*)(p + 16);   // k = 16+base..16+base+7
      af[i] = CAT16(lo, hi);
    }
#pragma unroll
    for (int i = 0; i < MI; ++i)
#pragma unroll
      for (int j = 0; j < NI; ++j)
        acc[i][j] = __builtin_amdgcn_wmma_f32_16x16x32_bf16(
            false, af[i], false, bfr[j], (short)0, acc[i][j], false, false);

    // commit next tile to the other buffer
    if (more) {
#ifndef USE_TDM
#pragma unroll
      for (int c = 0; c < ACH; ++c)
        *(v8bf*)&smem[(unsigned)(buf ^ 1) * AWORDS + (unsigned)arow * LDSTR + acol0 + c * 8] = aReg[c];
#endif
#pragma unroll
      for (int it = 0; it < (BN / 64); ++it) {
        int chunk = tid + it * 256;
        int r = chunk / (BN / 8);
        int c = (chunk % (BN / 8)) * 8;
#pragma unroll
        for (int e = 0; e < 8; ++e)
          smem[2 * AWORDS + (unsigned)(buf ^ 1) * BWORDS + (unsigned)(c + e) * LDSTR + r] = bReg[it][e];
      }
#ifdef USE_TDM
      if (tid < 32) __builtin_amdgcn_s_wait_tensorcnt(0);
#endif
    }
    __syncthreads();
  }

  // ---- epilogue (C layout: lane = column, VGPR v = row v + 8*hl)
#pragma unroll
  for (int i = 0; i < MI; ++i)
#pragma unroll
    for (int j = 0; j < NI; ++j)
#pragma unroll
      for (int v = 0; v < 8; ++v) {
        const int r = m0 + mBase + i * 16 + v + hl * 8;
        const int c = n0 + nBase + j * 16 + nc;
        const long idx = cOff + (long)r * ldc + c;
        float f = acc[i][j][v];
        if (bias)  f += bias[c];
        if (resid) f += resid[idx];
        if (relu)  f = fmaxf(f, 0.f);
        if (outF)  outF[idx] = f;
        if (outH)  outH[idx] = f2bf(f);
      }
}

// ---------------------------------------------------------------------------
// Causal flash attention, one (b,h) per blockIdx.y, 64 queries per WG
// (4 waves x 16 queries). Key blocks of 64, online softmax, fragment-burst
// WMMA for q*k^T and P*v. q/k/v: [B,H,S,DH] bf16. Output [BS,E] bf16.
// ---------------------------------------------------------------------------
__global__ __launch_bounds__(128)
void attn_kernel(const __bf16* __restrict__ q, const __bf16* __restrict__ k,
                 const __bf16* __restrict__ v, __bf16* __restrict__ outH) {
  constexpr int LK = 72;                     // padded LDS stride (bf16 units)
  __shared__ __bf16 Kt[64 * LK];             // [key][dh] row-major
  __shared__ __bf16 Vt[64 * LK];             // [dh][key] transposed
  __shared__ __bf16 Pt[4][16 * LK];          // per-wave P tile [row][key]

  const int z  = blockIdx.y;                 // b*H + h
  const int b  = z >> 4;
  const int h  = z & 15;
  const int q0 = blockIdx.x * 64;
  const int tid  = threadIdx.x;
  const int lane = tid & 31;
  const int wave = tid >> 5;
  const int hl = lane >> 4;
  const int nc = lane & 15;

  const long headOff = (long)z * Sc * DHc;
  const __bf16* qh = q + headOff;
  const __bf16* kh = k + headOff;
  const __bf16* vh = v + headOff;

  // q fragments for this wave's 16 rows (A-matrix layout, 2 K-steps)
  const int qm = q0 + wave * 16 + nc;
  v16bf qf[2];
#pragma unroll
  for (int ks = 0; ks < 2; ++ks) {
    const __bf16* p = qh + (long)qm * DHc + ks * 32 + hl * 8;
    v8bf lo = *(const v8bf*)p;
    v8bf hi = *(const v8bf*)(p + 16);
    qf[ks] = CAT16(lo, hi);
  }

  v8f O[4];
#pragma unroll
  for (int nt = 0; nt < 4; ++nt) O[nt] = (v8f){0, 0, 0, 0, 0, 0, 0, 0};
  float mrow[8], lrow[8];
#pragma unroll
  for (int r = 0; r < 8; ++r) { mrow[r] = -1e30f; lrow[r] = 0.f; }

  const int nkb = q0 / 64 + 1;               // causal key-block count (uniform)
  for (int kb = 0; kb < nkb; ++kb) {
    const int kbase = kb * 64;
    // load K (row-major) and V (transposed) tiles: 64x64 each
#pragma unroll
    for (int i = 0; i < 4; ++i) {
      int chunk = tid + i * 128;
      int r = chunk >> 3;
      int c = (chunk & 7) * 8;
      *(v8bf*)&Kt[r * LK + c] = *(const v8bf*)(kh + (long)(kbase + r) * DHc + c);
      v8bf vv = *(const v8bf*)(vh + (long)(kbase + r) * DHc + c);
#pragma unroll
      for (int e = 0; e < 8; ++e) Vt[(c + e) * LK + r] = vv[e];
    }
    __syncthreads();

    // ---- scores = q * k^T : preload all 8 K-fragments, then burst
    v16bf kf[4][2];
#pragma unroll
    for (int nt = 0; nt < 4; ++nt)
#pragma unroll
      for (int ks = 0; ks < 2; ++ks) {
        const __bf16* p = &Kt[(nt * 16 + nc) * LK + ks * 32 + hl * 8];
        v8bf lo = *(const v8bf*)p;
        v8bf hi = *(const v8bf*)(p + 16);
        kf[nt][ks] = CAT16(lo, hi);
      }
    v8f st[4];
#pragma unroll
    for (int nt = 0; nt < 4; ++nt) st[nt] = (v8f){0, 0, 0, 0, 0, 0, 0, 0};
#pragma unroll
    for (int ks = 0; ks < 2; ++ks)
#pragma unroll
      for (int nt = 0; nt < 4; ++nt)
        st[nt] = __builtin_amdgcn_wmma_f32_16x16x32_bf16(
            false, qf[ks], false, kf[nt][ks], (short)0, st[nt], false, false);

    // ---- online softmax (row = vr + 8*hl; reductions in 16-lane halves)
#pragma unroll
    for (int vr = 0; vr < 8; ++vr) {
      const int qrow = q0 + wave * 16 + vr + hl * 8;
      float sv[4];
      float bm = -1e30f;
#pragma unroll
      for (int nt = 0; nt < 4; ++nt) {
        const int kcol = kbase + nt * 16 + nc;
        float s = st[nt][vr] * 0.125f;       // DH^-0.5
        if (kcol > qrow) s = -1e30f;         // causal mask
        sv[nt] = s;
        bm = fmaxf(bm, s);
      }
#pragma unroll
      for (int off = 1; off < 16; off <<= 1) bm = fmaxf(bm, __shfl_xor(bm, off, 32));
      const float mn = fmaxf(mrow[vr], bm);
      const float alpha = __expf(mrow[vr] - mn);
      mrow[vr] = mn;
      float ps = 0.f;
#pragma unroll
      for (int nt = 0; nt < 4; ++nt) {
        float p = __expf(sv[nt] - mn);
        ps += p;
        Pt[wave][(vr + hl * 8) * LK + nt * 16 + nc] = f2bf(p);
      }
#pragma unroll
      for (int off = 1; off < 16; off <<= 1) ps += __shfl_xor(ps, off, 32);
      lrow[vr] = lrow[vr] * alpha + ps;
#pragma unroll
      for (int nt = 0; nt < 4; ++nt) O[nt][vr] *= alpha;
    }
    asm volatile("s_wait_dscnt 0" ::: "memory");   // P tile visible wave-wide

    // ---- O += P * V : preload P and V fragments, then burst
    v16bf pf[2];
#pragma unroll
    for (int ks = 0; ks < 2; ++ks) {
      const __bf16* p = &Pt[wave][nc * LK + ks * 32 + hl * 8];
      v8bf lo = *(const v8bf*)p;
      v8bf hi = *(const v8bf*)(p + 16);
      pf[ks] = CAT16(lo, hi);
    }
    v16bf vf[4][2];
#pragma unroll
    for (int nt = 0; nt < 4; ++nt)
#pragma unroll
      for (int ks = 0; ks < 2; ++ks) {
        const __bf16* p = &Vt[(nt * 16 + nc) * LK + ks * 32 + hl * 8];
        v8bf lo = *(const v8bf*)p;
        v8bf hi = *(const v8bf*)(p + 16);
        vf[nt][ks] = CAT16(lo, hi);
      }
#pragma unroll
    for (int ks = 0; ks < 2; ++ks)
#pragma unroll
      for (int nt = 0; nt < 4; ++nt)
        O[nt] = __builtin_amdgcn_wmma_f32_16x16x32_bf16(
            false, pf[ks], false, vf[nt][ks], (short)0, O[nt], false, false);
    __syncthreads();
  }

  // ---- normalize & write concatenated heads: out[(b*S+r)*E + h*64 + dh]
#pragma unroll
  for (int nt = 0; nt < 4; ++nt)
#pragma unroll
    for (int vr = 0; vr < 8; ++vr) {
      const int r = q0 + wave * 16 + vr + hl * 8;
      const float o = O[nt][vr] / lrow[vr];
      outH[((long)b * Sc + r) * Ec + h * 64 + nt * 16 + nc] = f2bf(o);
    }
}

// ---------------------------------------------------------------------------
// Host-side orchestration
// ---------------------------------------------------------------------------
extern "C" void kernel_launch(void* const* d_in, const int* in_sizes, int n_in,
                              void* d_out, int out_size, void* d_ws, size_t ws_size,
                              hipStream_t stream) {
  (void)in_sizes; (void)n_in; (void)out_size; (void)ws_size;
  const float* x     = (const float*)d_in[0];
  const float* Wq    = (const float*)d_in[1];
  const float* Wk    = (const float*)d_in[2];
  const float* Wv    = (const float*)d_in[3];
  const float* Wo    = (const float*)d_in[4];
  const float* bo    = (const float*)d_in[5];
  const float* W1    = (const float*)d_in[6];
  const float* b1    = (const float*)d_in[7];
  const float* W2    = (const float*)d_in[8];
  const float* b2    = (const float*)d_in[9];
  const float* g1    = (const float*)d_in[10];
  const float* beta1 = (const float*)d_in[11];
  const float* g2    = (const float*)d_in[12];
  const float* beta2 = (const float*)d_in[13];
  float* out = (float*)d_out;

  char* ws = (char*)d_ws;
  const size_t MB = 1ull << 20;
  __bf16* nx1H  = (__bf16*)(ws + 0);        // 8 MB, reused as nx2H
  __bf16* qH    = (__bf16*)(ws + 8 * MB);   // 8 MB
  __bf16* kH    = (__bf16*)(ws + 16 * MB);  // 8 MB
  __bf16* vH    = (__bf16*)(ws + 24 * MB);  // 8 MB
  __bf16* attnH = (__bf16*)(ws + 32 * MB);  // 8 MB
  __bf16* hH    = (__bf16*)(ws + 8 * MB);   // 32 MB, aliases q/k/v/attn (dead by FFN1)
  float*  x1F   = (float*)(ws + 40 * MB);   // 16 MB
  __bf16* WqH   = (__bf16*)(ws + 56 * MB);
  __bf16* WkH   = (__bf16*)(ws + 58 * MB);
  __bf16* WvH   = (__bf16*)(ws + 60 * MB);
  __bf16* WoH   = (__bf16*)(ws + 62 * MB);
  __bf16* W1H   = (__bf16*)(ws + 64 * MB);
  __bf16* W2H   = (__bf16*)(ws + 72 * MB);
  __bf16* nx2H  = nx1H;

  // --- weight conversion fp32 -> bf16
  const long nQKV = (long)Hc * Ec * DHc;
  f32_to_bf16_kernel<<<(unsigned)(nQKV / 1024), 256, 0, stream>>>(Wq, WqH, nQKV);
  f32_to_bf16_kernel<<<(unsigned)(nQKV / 1024), 256, 0, stream>>>(Wk, WkH, nQKV);
  f32_to_bf16_kernel<<<(unsigned)(nQKV / 1024), 256, 0, stream>>>(Wv, WvH, nQKV);
  const long nWo = (long)Ec * Ec;
  f32_to_bf16_kernel<<<(unsigned)(nWo / 1024), 256, 0, stream>>>(Wo, WoH, nWo);
  const long nW1 = (long)Ec * FFc;
  f32_to_bf16_kernel<<<(unsigned)(nW1 / 1024), 256, 0, stream>>>(W1, W1H, nW1);
  f32_to_bf16_kernel<<<(unsigned)(nW1 / 1024), 256, 0, stream>>>(W2, W2H, nW1);

  // --- LN1
  layernorm_kernel<<<BSc, 256, 0, stream>>>(x, g1, beta1, nx1H);

  // dynamic LDS sizes: double-buffered A + B tiles
  const size_t sh128 = (size_t)(2 * 128 * LDSTR + 2 * 128 * LDSTR) * 2;  // 40960
  const size_t sh64  = (size_t)(2 * 128 * LDSTR + 2 * 64 * LDSTR) * 2;   // 30720

  // --- QKV projections: per (b,h) batch, N=DH=64
  dim3 gq(1, Sc / 128, Bc * Hc);
  gemm_bf16<128, 64><<<gq, 256, sh64, stream>>>(
      nx1H, Ec, (long)Sc * Ec, WqH, DHc, (long)Ec * DHc,
      nullptr, nullptr, nullptr, qH, DHc, (long)Sc * DHc, Ec, Hc, 0, Sc);
  gemm_bf16<128, 64><<<gq, 256, sh64, stream>>>(
      nx1H, Ec, (long)Sc * Ec, WkH, DHc, (long)Ec * DHc,
      nullptr, nullptr, nullptr, kH, DHc, (long)Sc * DHc, Ec, Hc, 0, Sc);
  gemm_bf16<128, 64><<<gq, 256, sh64, stream>>>(
      nx1H, Ec, (long)Sc * Ec, WvH, DHc, (long)Ec * DHc,
      nullptr, nullptr, nullptr, vH, DHc, (long)Sc * DHc, Ec, Hc, 0, Sc);

  // --- causal flash attention
  attn_kernel<<<dim3(Sc / 64, Bc * Hc), 128, 0, stream>>>(qH, kH, vH, attnH);

  // --- Wo projection + bias + residual(x) -> x1 (fp32)
  gemm_bf16<128, 128><<<dim3(Ec / 128, BSc / 128, 1), 256, sh128, stream>>>(
      attnH, Ec, 0, WoH, Ec, 0, bo, x, x1F, nullptr, Ec, 0, Ec, 1, 0, BSc);

  // --- LN2
  layernorm_kernel<<<BSc, 256, 0, stream>>>(x1F, g2, beta2, nx2H);

  // --- FFN1: relu(nx2 @ W1 + b1) -> hH (bf16)
  gemm_bf16<128, 128><<<dim3(FFc / 128, BSc / 128, 1), 256, sh128, stream>>>(
      nx2H, Ec, 0, W1H, FFc, 0, b1, nullptr, nullptr, hH, FFc, 0, Ec, 1, 1, BSc);

  // --- FFN2: hH @ W2 + b2 + residual(x1) -> out (fp32)
  gemm_bf16<128, 128><<<dim3(Ec / 128, BSc / 128, 1), 256, sh128, stream>>>(
      hH, FFc, 0, W2H, Ec, 0, b2, x1F, out, nullptr, Ec, 0, FFc, 1, 0, BSc);
}
